// MultiheadMultiAttention_49091476193577
// MI455X (gfx1250) — matrix operand order repaired and checked by reading, compiled, not run on hardware
//
#include <hip/hip_runtime.h>
#include <hip/hip_bf16.h>
#include <stdint.h>

// ---------------------------------------------------------------------------
// MI455X (gfx1250): bf16 WMMA (16x16x32, f32 accum) for all GEMM-like stages,
// flash-style online softmax attention, async global->LDS staging (ASYNCcnt)
// with double-buffered LDS in the big GEMM.
// ---------------------------------------------------------------------------

typedef __attribute__((ext_vector_type(16))) __bf16 v16bf;
typedef __attribute__((ext_vector_type(8)))  __bf16 v8bf;
typedef __attribute__((ext_vector_type(8)))  float  v8f;

#define NSEQ 2048
#define BB   4
#define EE   1024
#define DD   64
#define SCALE 0.125f   // 64^-0.5

// --- fragment loaders (LDS or global, 16B-aligned) -------------------------
__device__ inline v16bf ld16s(const __bf16* p0, const __bf16* p1) {
  v8bf a = *(const v8bf*)p0;
  v8bf b = *(const v8bf*)p1;
  v16bf r;
#pragma unroll
  for (int i = 0; i < 8; ++i) { r[i] = a[i]; r[i + 8] = b[i]; }
  return r;
}
__device__ inline v16bf ld16c(const __bf16* p) { return ld16s(p, p + 8); }

__device__ inline v8f wmma_bf16(v16bf a, v16bf b, v8f c) {
  return __builtin_amdgcn_wmma_f32_16x16x32_bf16(
      /*neg_a=*/false, a, /*neg_b=*/false, b,
      /*c_mod=*/(short)0, c, /*reuse_a=*/false, /*reuse_b=*/false);
}

// --- gfx1250 async global->LDS copy (16B per lane), tracked by ASYNCcnt ----
__device__ inline void async_ld16(const __bf16* gptr, __bf16* lptr) {
  uint32_t loff = (uint32_t)(uintptr_t)(void*)lptr;  // low 32b == LDS offset
  asm volatile("global_load_async_to_lds_b128 %0, %1, off"
               :: "v"(loff), "v"(gptr) : "memory");
}
__device__ inline void wait_async0() {
  asm volatile("s_wait_asynccnt 0" ::: "memory");
}

// ---------------------------------------------------------------------------
// fp32 -> bf16 elementwise conversion
// ---------------------------------------------------------------------------
__global__ void cvt_bf16(const float* __restrict__ src, __bf16* __restrict__ dst,
                         int count) {
  int i = blockIdx.x * 256 + threadIdx.x;
  if (i < count) dst[i] = (__bf16)src[i];
}

// pack conv weight (D,D,KS) -> Wf[o][s*64+din], bf16
__global__ void pack_convw(const float* __restrict__ w, __bf16* __restrict__ dst) {
  int i = blockIdx.x * 256 + threadIdx.x;           // 64*64*4 = 16384
  if (i >= 64 * 64 * 4) return;
  int s = i & 3, din = (i >> 2) & 63, o = i >> 8;   // w[o][din][s] flat == i
  dst[o * 256 + s * 64 + din] = (__bf16)w[i];
}

// build Vt[(h*4+b)*64 + d][n] from Vb[(n*4+b)*1024 + h*64 + d], heads 0..11
__global__ void build_vt(const __bf16* __restrict__ Vb, __bf16* __restrict__ Vt) {
  size_t i = (size_t)blockIdx.x * 256 + threadIdx.x;  // 12*4*64*2048
  if (i >= 12ull * 4 * 64 * 2048) return;
  int n_i = (int)(i & 2047);
  size_t r = i >> 11;
  int d = (int)(r & 63); r >>= 6;
  int b_i = (int)(r & 3); int h = (int)(r >> 2);
  Vt[i] = Vb[((size_t)n_i * BB + b_i) * EE + h * DD + d];
}

// ---------------------------------------------------------------------------
// GEMM: C[m,n] = sum_k A[m,k] * W[n,k]   (A: MxK bf16, W: NxK bf16 row-major)
// mode 0: bf16 row-major to Cb ; mode 2: fp32 + bias to Cf
// block 256 threads (8 waves), tile M=256 N=64 Kc=32; wave tile 32x64.
// Double-buffered LDS, async global->LDS staging overlapped with WMMA.
// ---------------------------------------------------------------------------
__global__ __launch_bounds__(256) void gemm_xwT(
    const __bf16* __restrict__ A, const __bf16* __restrict__ W,
    const float* __restrict__ bias, __bf16* __restrict__ Cb,
    float* __restrict__ Cf, int K, int ldc, int mode) {
  __shared__ __align__(16) __bf16 As[2][256][32];
  __shared__ __align__(16) __bf16 Bs[2][64][32];
  const int t = threadIdx.x, lane = t & 31, wave = t >> 5;
  const int half = lane >> 4, l16 = lane & 15;
  const int m0 = blockIdx.x * 256, n0 = blockIdx.y * 64;

  auto stage = [&](int buf, int k0) {
    // A tile 256x32: 4 x 16B chunks per thread
#pragma unroll
    for (int c = 0; c < 4; ++c) {
      int cc = t * 4 + c;                   // 0..1023
      int r = cc >> 2, col = (cc & 3) * 8;
      async_ld16(A + (size_t)(m0 + r) * K + k0 + col, &As[buf][r][col]);
    }
    // B tile 64x32: 1 x 16B chunk per thread
    {
      int r = t >> 2, col = (t & 3) * 8;
      async_ld16(W + (size_t)(n0 + r) * K + k0 + col, &Bs[buf][r][col]);
    }
  };

  v8f acc[2][4] = {};
  const int nIter = K / 32;
  stage(0, 0);
  for (int i = 0; i < nIter; ++i) {
    const int cur = i & 1;
    wait_async0();          // own stage(cur) complete (async loads in-order)
    __syncthreads();        // all waves' stage(cur) visible; prev reads done
    if (i + 1 < nIter) stage(cur ^ 1, (i + 1) * 32);  // overlap with compute
    v16bf a0 = ld16c(&As[cur][wave * 32 + l16][16 * half]);
    v16bf a1 = ld16c(&As[cur][wave * 32 + 16 + l16][16 * half]);
#pragma unroll
    for (int c = 0; c < 4; ++c) {
      v16bf bfr = ld16c(&Bs[cur][16 * c + l16][16 * half]);
      acc[0][c] = wmma_bf16(a0, bfr, acc[0][c]);
      acc[1][c] = wmma_bf16(a1, bfr, acc[1][c]);
    }
  }
#pragma unroll
  for (int s = 0; s < 2; ++s) {
    const int mrow = m0 + wave * 32 + s * 16;
#pragma unroll
    for (int c = 0; c < 4; ++c) {
      int col = n0 + 16 * c + l16;
      float bv = (mode == 2) ? bias[col] : 0.f;
#pragma unroll
      for (int v = 0; v < 8; ++v) {
        int m = mrow + v + 8 * half;
        if (mode == 2) Cf[(size_t)m * ldc + col] = acc[s][c][v] + bv;
        else           Cb[(size_t)m * ldc + col] = (__bf16)acc[s][c][v];
      }
    }
  }
}

// ---------------------------------------------------------------------------
// Compressed-head conv as GEMM: y[t,o] = sum_{j<256} Xf[t,j]*Wf[o,j] + bias[o]
// Xf[t, s*64+din] = Xproj[(4t+s)*4 + b][hcol + din]
// block 128 threads (4 waves), tile M=64 N=64 Kc=32, K=256
// ---------------------------------------------------------------------------
__global__ __launch_bounds__(128) void conv_gemm(
    const __bf16* __restrict__ X, const __bf16* __restrict__ Wf,
    const float* __restrict__ bias, __bf16* __restrict__ out,
    int transpose_out) {
  const int hb = blockIdx.y;               // h'*4 + b
  const int b_i = hb & 3, hp = hb >> 2;
  const int hcol = (12 + hp) * DD;
  __shared__ __align__(16) __bf16 As[64][32];
  __shared__ __align__(16) __bf16 Bs[64][32];
  const int t = threadIdx.x, lane = t & 31, wave = t >> 5;
  const int half = lane >> 4, l16 = lane & 15;
  const int t0 = blockIdx.x * 64;
  v8f acc[4] = {};
  for (int k0 = 0; k0 < 256; k0 += 32) {
    {  // stage A 64x32 (16 elems / thread)
      int r = t >> 1, c = (t & 1) * 16;
#pragma unroll
      for (int cc = 0; cc < 16; cc += 8) {
        int j = k0 + c + cc;               // 8-aligned, never crosses s-block
        int s = j >> 6, din = j & 63;
        const __bf16* src =
            X + ((size_t)(4 * (t0 + r) + s) * BB + b_i) * EE + hcol + din;
        *(v8bf*)&As[r][c + cc] = *(const v8bf*)src;
      }
    }
    {  // stage B 64x32
      int r = t >> 1, c = (t & 1) * 16;
      const __bf16* src = Wf + (size_t)r * 256 + k0 + c;
      *(v8bf*)&Bs[r][c]     = *(const v8bf*)src;
      *(v8bf*)&Bs[r][c + 8] = *(const v8bf*)(src + 8);
    }
    __syncthreads();
    v16bf af = ld16c(&As[wave * 16 + l16][16 * half]);
#pragma unroll
    for (int c = 0; c < 4; ++c) {
      v16bf bfr = ld16c(&Bs[16 * c + l16][16 * half]);
      acc[c] = wmma_bf16(af, bfr, acc[c]);
    }
    __syncthreads();
  }
#pragma unroll
  for (int c = 0; c < 4; ++c) {
    int dout = 16 * c + l16;
    float bv = bias[dout];
#pragma unroll
    for (int v = 0; v < 8; ++v) {
      int tt = t0 + wave * 16 + v + 8 * half;
      float val = acc[c][v] + bv;
      if (transpose_out) out[((size_t)hb * 64 + dout) * 512 + tt] = (__bf16)val;
      else               out[((size_t)hb * 512 + tt) * 64 + dout] = (__bf16)val;
    }
  }
}

// ---------------------------------------------------------------------------
// Flash attention. block = 128 threads (4 waves), each wave one 16-query tile,
// block covers 64 queries (== one local window). Key loop in blocks of 32.
// mode 0: full (Nk=2048)  mode 1: local (window +/- look-back)
// mode 2: compressed (K/V from Kc/Vct, Nk=512, mask = AND of 4)
// K/V tiles staged with async global->LDS copies.
// ---------------------------------------------------------------------------
__global__ __launch_bounds__(128) void flash_attn(
    const __bf16* __restrict__ Qb, const __bf16* __restrict__ Ksrc,
    const __bf16* __restrict__ Vsrc, const unsigned char* __restrict__ maskkv,
    __bf16* __restrict__ Ob, int mode, int hbase) {
  const int hb = blockIdx.y, b_i = hb & 3, hl = hb >> 2;
  const int h = hbase + hl;
  const int q0 = blockIdx.x * 64;
  const int t = threadIdx.x, lane = t & 31, wave = t >> 5;
  const int half = lane >> 4, l16 = lane & 15;

  size_t kbase, vbase; int kjs, vds, nk0, nk1;
  if (mode == 2) {
    kbase = (size_t)hb * 512 * 64; kjs = 64;
    vbase = (size_t)hb * 64 * 512; vds = 512;
    nk0 = 0; nk1 = 512;
  } else {
    kbase = (size_t)b_i * EE + h * DD;          kjs = BB * EE;
    vbase = ((size_t)h * BB + b_i) * DD * NSEQ; vds = NSEQ;
    if (mode == 1) {
      int w = blockIdx.x;
      nk0 = (w == 0) ? 0 : (w * 64 - 64);
      nk1 = w * 64 + 64;
    } else { nk0 = 0; nk1 = NSEQ; }
  }

  __shared__ __align__(16) __bf16 Ks[32][64];
  __shared__ __align__(16) __bf16 Vs[64][32];
  __shared__ __align__(16) __bf16 Ps[4][16][32];

  // Q fragments (2 chunks of K=32 over D=64), held in registers for the loop
  const int qi = q0 + wave * 16 + l16;
  const __bf16* qrow = Qb + ((size_t)qi * BB + b_i) * EE + h * DD;
  v16bf qf0 = ld16c(qrow + 16 * half);
  v16bf qf1 = ld16c(qrow + 32 + 16 * half);

  v8f o0 = {}, o1 = {}, o2 = {}, o3 = {};
  float mrow[8], lrow[8];
#pragma unroll
  for (int v = 0; v < 8; ++v) { mrow[v] = -3.0e38f; lrow[v] = 0.f; }

  for (int j0 = nk0; j0 < nk1; j0 += 32) {
    {  // stage K tile 32x64 (row-major over d): 2 asyncs / thread
      int r = t >> 2, c = (t & 3) * 16;
      const __bf16* src = Ksrc + kbase + (size_t)(j0 + r) * kjs + c;
      async_ld16(src,     &Ks[r][c]);
      async_ld16(src + 8, &Ks[r][c + 8]);
    }
    {  // stage V^T tile 64x32 (d-major): 2 asyncs / thread
      int r = t >> 1, c = (t & 1) * 16;
      const __bf16* src = Vsrc + vbase + (size_t)r * vds + j0 + c;
      async_ld16(src,     &Vs[r][c]);
      async_ld16(src + 8, &Vs[r][c + 8]);
    }
    wait_async0();
    __syncthreads();

    // S = Q @ K^T : two 16-key chunks, each accumulating over d in 2 steps
    v8f s0 = {}, s1 = {};
    s0 = wmma_bf16(qf0, ld16c(&Ks[l16][16 * half]), s0);
    s0 = wmma_bf16(qf1, ld16c(&Ks[l16][32 + 16 * half]), s0);
    s1 = wmma_bf16(qf0, ld16c(&Ks[16 + l16][16 * half]), s1);
    s1 = wmma_bf16(qf1, ld16c(&Ks[16 + l16][32 + 16 * half]), s1);

    // key-pad mask (column-dependent only)
    float madd0 = 0.f, madd1 = 0.f;
    {
      int ja = j0 + l16, jb2 = j0 + 16 + l16;
      const unsigned char* mp = maskkv + (size_t)b_i * NSEQ;
      bool ma, mb;
      if (mode == 2) {
        ma = mp[4 * ja] && mp[4 * ja + 1] && mp[4 * ja + 2] && mp[4 * ja + 3];
        mb = mp[4 * jb2] && mp[4 * jb2 + 1] && mp[4 * jb2 + 2] && mp[4 * jb2 + 3];
      } else { ma = mp[ja] != 0; mb = mp[jb2] != 0; }
      madd0 = ma ? -1.0e9f : 0.f;
      madd1 = mb ? -1.0e9f : 0.f;
    }

    // online softmax (rows live on the 16 lanes of one half -> shfl_xor 1..8)
    float p0[8], p1[8], alpha[8];
#pragma unroll
    for (int v = 0; v < 8; ++v) {
      float a = s0[v] * SCALE + madd0;
      float b = s1[v] * SCALE + madd1;
      float rm = fmaxf(a, b);
      rm = fmaxf(rm, __shfl_xor(rm, 1));
      rm = fmaxf(rm, __shfl_xor(rm, 2));
      rm = fmaxf(rm, __shfl_xor(rm, 4));
      rm = fmaxf(rm, __shfl_xor(rm, 8));
      float mn = fmaxf(mrow[v], rm);
      float al = __expf(mrow[v] - mn);
      float e0 = __expf(a - mn), e1 = __expf(b - mn);
      float rs = e0 + e1;
      rs += __shfl_xor(rs, 1); rs += __shfl_xor(rs, 2);
      rs += __shfl_xor(rs, 4); rs += __shfl_xor(rs, 8);
      lrow[v] = lrow[v] * al + rs;
      mrow[v] = mn; alpha[v] = al; p0[v] = e0; p1[v] = e1;
    }
#pragma unroll
    for (int v = 0; v < 8; ++v) {
      o0[v] *= alpha[v]; o1[v] *= alpha[v];
      o2[v] *= alpha[v]; o3[v] *= alpha[v];
    }

    // P (C-layout) -> LDS -> A-layout fragment
#pragma unroll
    for (int v = 0; v < 8; ++v) {
      Ps[wave][v + 8 * half][l16]      = (__bf16)p0[v];
      Ps[wave][v + 8 * half][16 + l16] = (__bf16)p1[v];
    }
    asm volatile("s_wait_dscnt 0" ::: "memory");  // intra-wave LDS RAW

    v16bf pf = ld16c(&Ps[wave][l16][16 * half]);
    o0 = wmma_bf16(pf, ld16c(&Vs[l16][16 * half]), o0);
    o1 = wmma_bf16(pf, ld16c(&Vs[16 + l16][16 * half]), o1);
    o2 = wmma_bf16(pf, ld16c(&Vs[32 + l16][16 * half]), o2);
    o3 = wmma_bf16(pf, ld16c(&Vs[48 + l16][16 * half]), o3);

    __syncthreads();
  }

  // normalize and store to Ob (concat-head layout, bf16)
#pragma unroll
  for (int v = 0; v < 8; ++v) {
    float inv = 1.0f / lrow[v];
    int q2 = q0 + wave * 16 + v + 8 * half;
    size_t ro = ((size_t)q2 * BB + b_i) * EE + h * DD;
    Ob[ro + l16]      = (__bf16)(o0[v] * inv);
    Ob[ro + 16 + l16] = (__bf16)(o1[v] * inv);
    Ob[ro + 32 + l16] = (__bf16)(o2[v] * inv);
    Ob[ro + 48 + l16] = (__bf16)(o3[v] * inv);
  }
}

// ---------------------------------------------------------------------------
extern "C" void kernel_launch(void* const* d_in, const int* in_sizes, int n_in,
                              void* d_out, int out_size, void* d_ws, size_t ws_size,
                              hipStream_t stream) {
  const float* q    = (const float*)d_in[0];
  const float* k    = (const float*)d_in[1];
  const float* v    = (const float*)d_in[2];
  const unsigned char* mask_kv = (const unsigned char*)d_in[4]; // bool (b,n)
  const float* wq   = (const float*)d_in[5];
  const float* wk   = (const float*)d_in[6];
  const float* wv   = (const float*)d_in[7];
  const float* wo   = (const float*)d_in[8];
  const float* wo_b = (const float*)d_in[9];
  const float* ckw  = (const float*)d_in[10];
  const float* ckb  = (const float*)d_in[11];
  const float* cvw  = (const float*)d_in[12];
  const float* cvb  = (const float*)d_in[13];
  float* out = (float*)d_out;

  // workspace carve (256B aligned)
  size_t off = 0;
  auto carve = [&](size_t bytes) {
    void* p = (char*)d_ws + off;
    off += (bytes + 255) & ~(size_t)255;
    return p;
  };
  const size_t XB = (size_t)NSEQ * BB * EE * 2;   // 16 MB
  __bf16* Xq = (__bf16*)carve(XB);
  __bf16* Xk = (__bf16*)carve(XB);
  __bf16* Xv = (__bf16*)carve(XB);
  __bf16* Wq = (__bf16*)carve((size_t)EE * EE * 2);
  __bf16* Wk = (__bf16*)carve((size_t)EE * EE * 2);
  __bf16* Wv = (__bf16*)carve((size_t)EE * EE * 2);
  __bf16* Wo = (__bf16*)carve((size_t)EE * EE * 2);
  __bf16* Wck = (__bf16*)carve(64 * 256 * 2);
  __bf16* Wcv = (__bf16*)carve(64 * 256 * 2);
  __bf16* Qb = (__bf16*)carve(XB);
  __bf16* Kb = (__bf16*)carve(XB);
  __bf16* Vb = (__bf16*)carve(XB);
  __bf16* Vt = (__bf16*)carve(12ull * 4 * 64 * NSEQ * 2);
  __bf16* Kc  = (__bf16*)carve(16ull * 512 * 64 * 2);
  __bf16* Vct = (__bf16*)carve(16ull * 512 * 64 * 2);
  __bf16* Ob = (__bf16*)carve(XB);

  const int nx = NSEQ * BB * EE;  // 8388608
  const int nw = EE * EE;         // 1048576

  // 1) convert inputs + weights to bf16
  cvt_bf16<<<(nx + 255) / 256, 256, 0, stream>>>(q, Xq, nx);
  cvt_bf16<<<(nx + 255) / 256, 256, 0, stream>>>(k, Xk, nx);
  cvt_bf16<<<(nx + 255) / 256, 256, 0, stream>>>(v, Xv, nx);
  cvt_bf16<<<(nw + 255) / 256, 256, 0, stream>>>(wq, Wq, nw);
  cvt_bf16<<<(nw + 255) / 256, 256, 0, stream>>>(wk, Wk, nw);
  cvt_bf16<<<(nw + 255) / 256, 256, 0, stream>>>(wv, Wv, nw);
  cvt_bf16<<<(nw + 255) / 256, 256, 0, stream>>>(wo, Wo, nw);
  pack_convw<<<64, 256, 0, stream>>>(ckw, Wck);
  pack_convw<<<64, 256, 0, stream>>>(cvw, Wcv);

  // 2) projections: Qb/Kb/Vb = X @ W^T  (M=8192, N=K=1024)
  dim3 gg(NSEQ * BB / 256, EE / 64);
  gemm_xwT<<<gg, 256, 0, stream>>>(Xq, Wq, nullptr, Qb, nullptr, EE, EE, 0);
  gemm_xwT<<<gg, 256, 0, stream>>>(Xk, Wk, nullptr, Kb, nullptr, EE, EE, 0);
  gemm_xwT<<<gg, 256, 0, stream>>>(Xv, Wv, nullptr, Vb, nullptr, EE, EE, 0);

  // 3) V^T for heads 0..11 (full + local)
  build_vt<<<(int)((12ull * 4 * 64 * NSEQ + 255) / 256), 256, 0, stream>>>(Vb, Vt);

  // 4) compressed K/V via conv-as-GEMM (M=512, N=64, K=256 per (h',b))
  dim3 gc(512 / 64, 16);
  conv_gemm<<<gc, 128, 0, stream>>>(Kb, Wck, ckb, Kc, 0);
  conv_gemm<<<gc, 128, 0, stream>>>(Vb, Wcv, cvb, Vct, 1);

  // 5) attention (flash): full heads 0..7, local 8..11, compressed 12..15
  flash_attn<<<dim3(NSEQ / 64, 8 * BB), 128, 0, stream>>>(Qb, Kb, Vt, mask_kv, Ob, 0, 0);
  flash_attn<<<dim3(NSEQ / 64, 4 * BB), 128, 0, stream>>>(Qb, Kb, Vt, mask_kv, Ob, 1, 8);
  flash_attn<<<dim3(NSEQ / 64, 4 * BB), 128, 0, stream>>>(Qb, Kc, Vct, mask_kv, Ob, 2, 12);

  // 6) output projection: out = Ob @ Wo^T + wo_b  (fp32, (n,b,E) row == m)
  gemm_xwT<<<gg, 256, 0, stream>>>(Ob, Wo, wo_b, nullptr, out, EE, EE, 2);

  (void)in_sizes; (void)n_in; (void)out_size; (void)ws_size;
}
